// RWKV6TimeMix_34883724378693
// MI455X (gfx1250) — matrix operand order repaired and checked by reading, compiled, not run on hardware
//
#include <hip/hip_runtime.h>
#include <hip/hip_bf16.h>

// ---------------------------------------------------------------------------
// RWKV6 time-mix for MI455X (gfx1250, wave32, WMMA bf16 16x16x32)
//   - all big GEMMs: bf16 WMMA, 128x128 tiles, A/B staged to LDS with
//     GLOBAL_LOAD_ASYNC_TO_LDS_B128 (ASYNCcnt) -- CDNA5 async path
//   - weights pre-transposed to [Npad][K] bf16 (zero-padded) so the GEMM
//     hot loop has zero bounds checks and fully coalesced 128b transfers
//   - WKV: one block per (b,h), 64x64 state + all derived decay matrices
//     resident in ~153KB LDS (CDNA5 320KB WGP LDS), 4 WMMA matmuls/chunk
// ---------------------------------------------------------------------------

typedef __attribute__((ext_vector_type(16))) __bf16 v16bf;
typedef __attribute__((ext_vector_type(8)))  float  v8f;

union V16BF { v16bf v; unsigned int u[8]; unsigned short h[16]; };

__device__ __forceinline__ unsigned short f2bf(float f) {
  unsigned int u = __float_as_uint(f);
  u += 0x7FFFu + ((u >> 16) & 1u);           // round-to-nearest-even
  return (unsigned short)(u >> 16);
}
__device__ __forceinline__ float bf2f(unsigned short h) {
  return __uint_as_float(((unsigned int)h) << 16);
}
__device__ __forceinline__ v8f vzero8() {
  v8f z;
#pragma unroll
  for (int i = 0; i < 8; ++i) z[i] = 0.0f;
  return z;
}

// CDNA5 async global->LDS DMA (16B per lane), tracked by ASYNCcnt.
__device__ __forceinline__ void async_load_b128(const void* gptr, void* lptr) {
  unsigned lds = (unsigned)(size_t)lptr;     // low 32 bits of flat addr = LDS offset
  asm volatile("global_load_async_to_lds_b128 %0, %1, off"
               :: "v"(lds), "v"(gptr) : "memory");
}
__device__ __forceinline__ void wait_async0() {
  asm volatile("s_wait_asynccnt 0" ::: "memory");
}

// A fragment: 16x32 bf16, row-major source, leading dim `ld` (elements).
// lanes 0-15 = rows; half selects K 0..15 vs 16..31 (contiguous order, not ISA table order; B matches).
__device__ __forceinline__ V16BF load_fragA(const unsigned short* p, int ld, int row0, int k0) {
  int lane = threadIdx.x & 31;
  int m = lane & 15, half = lane >> 4;
  const unsigned short* base = p + (size_t)(row0 + m) * ld + k0 + half * 16;
  V16BF f;
#pragma unroll
  for (int i = 0; i < 8; ++i) f.u[i] = *(const unsigned int*)(base + i * 2);
  return f;
}
// B fragment: 32x16 bf16 from transposed storage p[n][k] (leading dim ld).
// Lanes 0-15 hold K=0..15 of column n, lanes 16-31 hold K=16..31.
__device__ __forceinline__ V16BF load_fragB(const unsigned short* p, int ld, int n0, int k0) {
  int lane = threadIdx.x & 31;
  int n = lane & 15, half = lane >> 4;
  const unsigned short* base = p + (size_t)(n0 + n) * ld + k0 + half * 16;
  V16BF f;
#pragma unroll
  for (int i = 0; i < 8; ++i) f.u[i] = *(const unsigned int*)(base + i * 2);
  return f;
}
__device__ __forceinline__ v8f wmma_bf16(const V16BF& a, const V16BF& b, v8f c) {
  return __builtin_amdgcn_wmma_f32_16x16x32_bf16(false, a.v, false, b.v, (short)0, c, false, false);
}

// ---------------------------------------------------------------------------
// Weight transpose+convert: src f32 [K][N] -> dst bf16 [Npad][K], pad rows = 0.
__global__ void k_transpose_bf16(const float* __restrict__ s, unsigned short* __restrict__ d,
                                 int K, int N, int Npad) {
  int i = blockIdx.x * 256 + threadIdx.x;
  if (i >= Npad * K) return;
  int n = i / K, k = i % K;
  d[i] = (n < N) ? f2bf(s[(size_t)k * N + n]) : (unsigned short)0;
}

__global__ void k_prep(const float* __restrict__ x, const float* __restrict__ maa_x,
                       float* __restrict__ xx, unsigned short* __restrict__ xmix,
                       int T, int D, int n) {
  int i = blockIdx.x * 256 + threadIdx.x;
  if (i >= n) return;
  int d = i % D;
  int t = (i / D) % T;
  float xv = x[i];
  float prev = (t > 0) ? x[i - D] : 0.0f;
  float xxv = prev - xv;                      // causal_shift(x) - x
  xx[i] = xxv;
  xmix[i] = f2bf(xv + xxv * maa_x[d]);
}

// ---------------------------------------------------------------------------
// bf16 WMMA GEMM: C[M,N] = A[M,K] * B[K,N] with B given TRANSPOSED as
// Bt[Npad][K] (zero-padded rows). Fused epilogues.
#define GM_F32       0
#define GM_F32_TANH  1
#define GM_BF16      2
#define GM_BF16_TANH 3
#define GM_F32_SILU  4
#define GM_WLOG      5

__global__ __launch_bounds__(256) void k_gemm_bf16(
    const unsigned short* __restrict__ A, const unsigned short* __restrict__ BT,
    float* __restrict__ outF, unsigned short* __restrict__ outB,
    const float* __restrict__ bias, int M, int N, int K, int mode) {
  __shared__ unsigned short As[128 * 32];
  __shared__ unsigned short Bs[128 * 32];   // Bs[n][k]
  int tid = threadIdx.x;
  int wid = tid >> 5, lane = tid & 31;
  int wm = wid >> 2, wn = wid & 3;          // waves: 2 (M) x 4 (N)
  int m0 = blockIdx.y * 128, n0 = blockIdx.x * 128;

  v8f acc[4][2];
#pragma unroll
  for (int i = 0; i < 4; ++i)
#pragma unroll
    for (int j = 0; j < 2; ++j) acc[i][j] = vzero8();

  int row = tid >> 1, col = (tid & 1) * 16; // identical 128x32 tile mapping for A and B
  const unsigned short* aBase = A  + (size_t)(m0 + row) * K + col;
  const unsigned short* bBase = BT + (size_t)(n0 + row) * K + col;

  for (int kk = 0; kk < K; kk += 32) {
    async_load_b128(aBase + kk,     &As[row * 32 + col]);
    async_load_b128(aBase + kk + 8, &As[row * 32 + col + 8]);
    async_load_b128(bBase + kk,     &Bs[row * 32 + col]);
    async_load_b128(bBase + kk + 8, &Bs[row * 32 + col + 8]);
    if (kk + 32 < K) {                       // prefetch next K-step
      __builtin_prefetch(aBase + kk + 32, 0, 1);
      __builtin_prefetch(bBase + kk + 32, 0, 1);
    }
    wait_async0();
    __syncthreads();
    V16BF bf0 = load_fragB(Bs, 32, wn * 32, 0);
    V16BF bf1 = load_fragB(Bs, 32, wn * 32 + 16, 0);
#pragma unroll
    for (int mt = 0; mt < 4; ++mt) {
      V16BF af = load_fragA(As, 32, wm * 64 + mt * 16, 0);
      acc[mt][0] = wmma_bf16(af, bf0, acc[mt][0]);
      acc[mt][1] = wmma_bf16(af, bf1, acc[mt][1]);
    }
    __syncthreads();
  }

  int half = lane >> 4, nn = lane & 15;
  const float LOGPREC = -5.298317366548036f; // ln(0.005)
#pragma unroll
  for (int mt = 0; mt < 4; ++mt)
#pragma unroll
    for (int nt = 0; nt < 2; ++nt)
#pragma unroll
      for (int r = 0; r < 8; ++r) {
        int gm = m0 + wm * 64 + mt * 16 + half * 8 + r;
        int gn = n0 + wn * 32 + nt * 16 + nn;
        if (gn >= N) continue;
        float v = acc[mt][nt][r];
        size_t o = (size_t)gm * N + gn;
        switch (mode) {
          case GM_F32:       outF[o] = v; break;
          case GM_F32_TANH:  outF[o] = tanhf(v); break;
          case GM_BF16:      outB[o] = f2bf(v); break;
          case GM_BF16_TANH: outB[o] = f2bf(tanhf(v)); break;
          case GM_F32_SILU:  outF[o] = v / (1.0f + expf(-v)); break;
          case GM_WLOG: {
            float w = -expf(bias[gn] + v);     // w = -exp(ww)
            outF[o] = fmaxf(w, LOGPREC);       // log(clip(exp(w), 0.005))
            break;
          }
        }
      }
}

// ---------------------------------------------------------------------------
// Rank-32 token-mix: m_f = xxx[t, f*32:(f+1)*32] @ w2[f] then xw..xg.
__global__ __launch_bounds__(256) void k_mix(
    const float* __restrict__ x, const float* __restrict__ xx, const float* __restrict__ xxx,
    const float* __restrict__ w2,
    const float* __restrict__ mw, const float* __restrict__ mk, const float* __restrict__ mv,
    const float* __restrict__ mr, const float* __restrict__ mg,
    unsigned short* __restrict__ xw, unsigned short* __restrict__ xk, unsigned short* __restrict__ xvp,
    unsigned short* __restrict__ xr, unsigned short* __restrict__ xg, int D) {
  int t = blockIdx.y;
  int d = blockIdx.x * 256 + threadIdx.x;
  __shared__ float sx[160];
  if (threadIdx.x < 160) sx[threadIdx.x] = xxx[(size_t)t * 160 + threadIdx.x];
  __syncthreads();
  float m[5];
#pragma unroll
  for (int f = 0; f < 5; ++f) {
    float acc = 0.0f;
#pragma unroll
    for (int i = 0; i < 32; ++i)
      acc += sx[f * 32 + i] * w2[(size_t)(f * 32 + i) * D + d];
    m[f] = acc;
  }
  size_t g = (size_t)t * D + d;
  float xv = x[g], xxv = xx[g];
  xw[g]  = f2bf(xv + xxv * (mw[d] + m[0]));
  xk[g]  = f2bf(xv + xxv * (mk[d] + m[1]));
  xvp[g] = f2bf(xv + xxv * (mv[d] + m[2]));
  xr[g]  = f2bf(xv + xxv * (mr[d] + m[3]));
  xg[g]  = f2bf(xv + xxv * (mg[d] + m[4]));
}

// ---------------------------------------------------------------------------
// Chunked WKV: one block per (b,h); 16 chunks of Tc=128 sequential; state in LDS.
__global__ __launch_bounds__(256) void k_wkv(
    const unsigned short* __restrict__ gR, const unsigned short* __restrict__ gK,
    const unsigned short* __restrict__ gV, const float* __restrict__ gWlog,
    const float* __restrict__ faaaa, float* __restrict__ att,
    int T, int D, int H) {
  const int b = blockIdx.x / H;
  const int h = blockIdx.x % H;
  const int tid = threadIdx.x;
  const int wid = tid >> 5, lane = tid & 31;
  const int half = lane >> 4, nn = lane & 15;

  __shared__ unsigned short sRa[128 * 64];   // r * exp(shifted - off)
  __shared__ unsigned short sKb[128 * 64];   // k * exp(off - cum)  (rows t, used as B^T)
  __shared__ unsigned short sRw[128 * 64];   // r * w_intra
  __shared__ unsigned short sKwT[64 * 128];  // (k * w_inter)^T : [kf][t]
  __shared__ unsigned short sVbT[64 * 128];  // v^T : [vdim][t]
  __shared__ unsigned short sSbT[64 * 64];   // state^T bf16 : [vdim][kf]
  __shared__ float sState[64 * 64];          // state[kf][vdim] f32
  __shared__ float sWkv[64 * 64];
  __shared__ float sWsum[64];
  __shared__ float sOff[64];
  __shared__ float sDiag[128];
  __shared__ float sU[64];
  __shared__ float sUnion[8192];             // 32KB: wcum f32[128][64] ALIAS Abf bf16[128][128]
  float* sWcum = sUnion;
  unsigned short* sAbf = (unsigned short*)sUnion;

  for (int i = tid; i < 4096; i += 256) sState[i] = 0.0f;
  if (tid < 64) sU[tid] = faaaa[h * 64 + tid];
  __syncthreads();

  const int NCH = T / 128;
  for (int n = 0; n < NCH; ++n) {
    const size_t rowBase = ((size_t)(b * T + n * 128)) * D + (size_t)h * 64;

    // Phase 1: per-column cumsum of wlog (serial over t, 64 columns)
    if (tid < 64) {
      int k = tid;
      float cum = 0.0f;
      for (int t = 0; t < 128; ++t) {
        cum += gWlog[rowBase + (size_t)t * D + k];
        sWcum[t * 64 + k] = cum;
        if (t == 63) sOff[k] = cum;           // off = shifted[Tc/2] = wcum[63]
      }
      sWsum[k] = cum;
    }
    if (tid < 128) sDiag[tid] = 0.0f;
    __syncthreads();

    // Phase 2: build derived bf16 matrices + diag + state snapshot
    for (int idx = tid; idx < 8192; idx += 256) {
      int t = idx >> 6, k = idx & 63;
      size_t g = rowBase + (size_t)t * D + k;
      float rv = bf2f(gR[g]), kv = bf2f(gK[g]), vv = bf2f(gV[g]);
      float cum = sWcum[idx];
      float wl  = gWlog[g];
      float shifted = cum - wl;
      float off = sOff[k];
      sRa[t * 64 + k]   = f2bf(rv * expf(shifted - off));
      sKb[t * 64 + k]   = f2bf(kv * expf(off - cum));
      sRw[t * 64 + k]   = f2bf(rv * expf(shifted));           // w_intra
      sKwT[k * 128 + t] = f2bf(kv * expf(sWsum[k] - cum));    // w_inter
      sVbT[k * 128 + t] = f2bf(vv);                           // here k indexes vdim
      atomicAdd(&sDiag[t], rv * sU[k] * kv);
    }
    for (int idx = tid; idx < 4096; idx += 256) {
      int kf = idx >> 6, vd = idx & 63;
      sSbT[vd * 64 + kf] = f2bf(sState[kf * 64 + vd]);
    }
    __syncthreads();

    // Phase 3: a = Ra @ Kb^T  (128x128), mask tril(-1) + diag -> sAbf (bf16)
    {
      v8f acc[8];
#pragma unroll
      for (int i = 0; i < 8; ++i) acc[i] = vzero8();
#pragma unroll
      for (int ks = 0; ks < 64; ks += 32) {
        V16BF af = load_fragA(sRa, 64, wid * 16, ks);
#pragma unroll
        for (int nt = 0; nt < 8; ++nt) {
          V16BF bf = load_fragB(sKb, 64, nt * 16, ks);
          acc[nt] = wmma_bf16(af, bf, acc[nt]);
        }
      }
#pragma unroll
      for (int nt = 0; nt < 8; ++nt)
#pragma unroll
        for (int r = 0; r < 8; ++r) {
          int i = wid * 16 + half * 8 + r;
          int j = nt * 16 + nn;
          float v = (i > j) ? acc[nt][r] : ((i == j) ? sDiag[i] : 0.0f);
          sAbf[i * 128 + j] = f2bf(v);
        }
    }
    __syncthreads();

    // Phase 4: out = a @ V + Rw @ state   (128x64) -> global att
    {
      v8f acc[4];
#pragma unroll
      for (int i = 0; i < 4; ++i) acc[i] = vzero8();
#pragma unroll
      for (int ks = 0; ks < 128; ks += 32) {
        V16BF af = load_fragA(sAbf, 128, wid * 16, ks);
#pragma unroll
        for (int nt = 0; nt < 4; ++nt) {
          V16BF bf = load_fragB(sVbT, 128, nt * 16, ks);
          acc[nt] = wmma_bf16(af, bf, acc[nt]);
        }
      }
#pragma unroll
      for (int ks = 0; ks < 64; ks += 32) {
        V16BF af = load_fragA(sRw, 64, wid * 16, ks);
#pragma unroll
        for (int nt = 0; nt < 4; ++nt) {
          V16BF bf = load_fragB(sSbT, 64, nt * 16, ks);
          acc[nt] = wmma_bf16(af, bf, acc[nt]);
        }
      }
#pragma unroll
      for (int nt = 0; nt < 4; ++nt)
#pragma unroll
        for (int r = 0; r < 8; ++r) {
          int i = wid * 16 + half * 8 + r;
          int j = nt * 16 + nn;
          att[rowBase + (size_t)i * D + j] = acc[nt][r];
        }
    }

    // Phase 5: wkv = Kw^T @ V  (64x64) -> sWkv
    {
      int mt = wid >> 1;
      int ntb = (wid & 1) * 2;
      v8f acc[2];
      acc[0] = vzero8(); acc[1] = vzero8();
#pragma unroll
      for (int ks = 0; ks < 128; ks += 32) {
        V16BF af = load_fragA(sKwT, 128, mt * 16, ks);
#pragma unroll
        for (int q = 0; q < 2; ++q) {
          V16BF bf = load_fragB(sVbT, 128, (ntb + q) * 16, ks);
          acc[q] = wmma_bf16(af, bf, acc[q]);
        }
      }
#pragma unroll
      for (int q = 0; q < 2; ++q)
#pragma unroll
        for (int r = 0; r < 8; ++r) {
          int ki = mt * 16 + half * 8 + r;
          int vj = (ntb + q) * 16 + nn;
          sWkv[ki * 64 + vj] = acc[q][r];
        }
    }
    __syncthreads();

    // Phase 6: state = state * exp(wsum[kf]) + wkv
    for (int idx = tid; idx < 4096; idx += 256) {
      int kf = idx >> 6;
      sState[idx] = sState[idx] * expf(sWsum[kf]) + sWkv[idx];
    }
    __syncthreads();
  }
}

// ---------------------------------------------------------------------------
// Per-head LayerNorm (over K=64) then * ln_w + ln_b, gate with g -> bf16
__global__ __launch_bounds__(64) void k_ln_gate(
    const float* __restrict__ att, const float* __restrict__ gf,
    const float* __restrict__ ln_w, const float* __restrict__ ln_b,
    unsigned short* __restrict__ gated, int D) {
  int h = blockIdx.x, t = blockIdx.y, k = threadIdx.x;
  __shared__ float red[64];
  size_t g = (size_t)t * D + h * 64 + k;
  float v = att[g];
  red[k] = v; __syncthreads();
  for (int s = 32; s > 0; s >>= 1) { if (k < s) red[k] += red[k + s]; __syncthreads(); }
  float mu = red[0] * (1.0f / 64.0f); __syncthreads();
  float dv = v - mu;
  red[k] = dv * dv; __syncthreads();
  for (int s = 32; s > 0; s >>= 1) { if (k < s) red[k] += red[k + s]; __syncthreads(); }
  float var = red[0] * (1.0f / 64.0f);
  float og = dv * rsqrtf(var + 6.4e-4f);     // LN_EPS = 1e-5 * 64
  int d = h * 64 + k;
  gated[g] = f2bf((og * ln_w[d] + ln_b[d]) * gf[g]);
}

// ---------------------------------------------------------------------------
extern "C" void kernel_launch(void* const* d_in, const int* in_sizes, int n_in,
                              void* d_out, int out_size, void* d_ws, size_t ws_size,
                              hipStream_t stream) {
  (void)in_sizes; (void)n_in; (void)out_size; (void)ws_size;
  const int Bn = 4, Tn = 2048, Dn = 1024, Hn = 16;
  const int M = Bn * Tn;                       // 8192
  const size_t MD = (size_t)M * Dn;

  const float* x      = (const float*)d_in[0];
  const float* maa_x  = (const float*)d_in[1];
  const float* maa_w  = (const float*)d_in[2];
  const float* maa_k  = (const float*)d_in[3];
  const float* maa_v  = (const float*)d_in[4];
  const float* maa_r  = (const float*)d_in[5];
  const float* maa_g  = (const float*)d_in[6];
  const float* w1     = (const float*)d_in[7];   // (1024,160)
  const float* w2     = (const float*)d_in[8];   // (5,32,1024)
  const float* tdec   = (const float*)d_in[9];   // (D,)
  const float* dw1    = (const float*)d_in[10];  // (1024,64)
  const float* dw2    = (const float*)d_in[11];  // (64,1024)
  const float* faaaa  = (const float*)d_in[12];  // (16,64)
  const float* Wr     = (const float*)d_in[13];
  const float* Wk     = (const float*)d_in[14];
  const float* Wv     = (const float*)d_in[15];
  const float* Wg     = (const float*)d_in[16];
  const float* Wo     = (const float*)d_in[17];
  const float* ln_w   = (const float*)d_in[18];
  const float* ln_b   = (const float*)d_in[19];
  float* out = (float*)d_out;

  char* ws = (char*)d_ws;
  size_t off = 0;
  auto take = [&](size_t bytes) -> char* {
    char* p = ws + off;
    off = (off + bytes + 255) & ~(size_t)255;
    return p;
  };
  // transposed bf16 weights: [Npad][K], zero-padded rows
  unsigned short* tW1  = (unsigned short*)take((size_t)256 * 1024 * 2);   // N=160 ->256
  unsigned short* tWr  = (unsigned short*)take((size_t)1024 * 1024 * 2);
  unsigned short* tWk  = (unsigned short*)take((size_t)1024 * 1024 * 2);
  unsigned short* tWv  = (unsigned short*)take((size_t)1024 * 1024 * 2);
  unsigned short* tWg  = (unsigned short*)take((size_t)1024 * 1024 * 2);
  unsigned short* tWo  = (unsigned short*)take((size_t)1024 * 1024 * 2);
  unsigned short* tdw1 = (unsigned short*)take((size_t)128 * 1024 * 2);   // N=64 ->128
  unsigned short* tdw2 = (unsigned short*)take((size_t)1024 * 64 * 2);
  float*          xxB  = (float*)take(MD * 4);
  unsigned short* xmix = (unsigned short*)take(MD * 2);
  float*          xxx  = (float*)take((size_t)M * 160 * 4);
  unsigned short* xwB  = (unsigned short*)take(MD * 2);
  unsigned short* xkB  = (unsigned short*)take(MD * 2);
  unsigned short* xvB  = (unsigned short*)take(MD * 2);
  unsigned short* xrB  = (unsigned short*)take(MD * 2);
  unsigned short* xgB  = (unsigned short*)take(MD * 2);
  unsigned short* t1B  = (unsigned short*)take((size_t)M * 64 * 2);
  float*          wlog = (float*)take(MD * 4);
  unsigned short* rB   = (unsigned short*)take(MD * 2);
  unsigned short* kB   = (unsigned short*)take(MD * 2);
  unsigned short* vB   = (unsigned short*)take(MD * 2);
  float*          gF   = (float*)take(MD * 4);
  float*          attF = (float*)take(MD * 4);
  unsigned short* gatd = (unsigned short*)take(MD * 2);

  auto tp = [&](const float* s, unsigned short* d, int K, int N, int Npad) {
    int n = Npad * K;
    k_transpose_bf16<<<(n + 255) / 256, 256, 0, stream>>>(s, d, K, N, Npad);
  };
  tp(w1, tW1, 1024, 160, 256);
  tp(Wr, tWr, 1024, 1024, 1024);
  tp(Wk, tWk, 1024, 1024, 1024);
  tp(Wv, tWv, 1024, 1024, 1024);
  tp(Wg, tWg, 1024, 1024, 1024);
  tp(Wo, tWo, 1024, 1024, 1024);
  tp(dw1, tdw1, 1024, 64, 128);
  tp(dw2, tdw2, 64, 1024, 1024);

  k_prep<<<(int)((MD + 255) / 256), 256, 0, stream>>>(x, maa_x, xxB, xmix, Tn, Dn, (int)MD);

  dim3 blk(256);
  auto gemm = [&](const unsigned short* A, const unsigned short* BT, float* oF,
                  unsigned short* oB, const float* bias, int Mm, int Nn, int Kk, int mode) {
    dim3 grid((Nn + 127) / 128, (Mm + 127) / 128);
    k_gemm_bf16<<<grid, blk, 0, stream>>>(A, BT, oF, oB, bias, Mm, Nn, Kk, mode);
  };

  gemm(xmix, tW1, xxx, nullptr, nullptr, M, 160, 1024, GM_F32_TANH);

  k_mix<<<dim3(Dn / 256, M), 256, 0, stream>>>(x, xxB, xxx, w2, maa_w, maa_k, maa_v,
                                               maa_r, maa_g, xwB, xkB, xvB, xrB, xgB, Dn);

  gemm(xrB, tWr, nullptr, rB, nullptr, M, Dn, Dn, GM_BF16);
  gemm(xkB, tWk, nullptr, kB, nullptr, M, Dn, Dn, GM_BF16);
  gemm(xvB, tWv, nullptr, vB, nullptr, M, Dn, Dn, GM_BF16);
  gemm(xgB, tWg, gF, nullptr, nullptr, M, Dn, Dn, GM_F32_SILU);
  gemm(xwB, tdw1, nullptr, t1B, nullptr, M, 64, 1024, GM_BF16_TANH);
  gemm(t1B, tdw2, wlog, nullptr, tdec, M, Dn, 64, GM_WLOG);

  k_wkv<<<Bn * Hn, 256, 0, stream>>>(rB, kB, vB, wlog, faaaa, attF, Tn, Dn, Hn);

  k_ln_gate<<<dim3(Hn, M), 64, 0, stream>>>(attF, gF, ln_w, ln_b, gatd, Dn);

  gemm(gatd, tWo, out, nullptr, nullptr, M, Dn, Dn, GM_F32);
}
